// self_attention1d_68702296867130
// MI455X (gfx1250) — matrix operand order repaired and checked by reading, not compiled, not run on hardware
//
#include <hip/hip_runtime.h>

// ---------------------------------------------------------------------------
// Fused QKV "self_attention1d" for MI455X (gfx1250, wave32, WMMA).
//
//   q = x@Wq.T + bq ; k,v likewise
//   kv[b] = sum_j k*v ; ksum[b] = sum_j |k| ; out = q*kv/(1+|q|*ksum)
//
// Compute bound (0.82 TFLOP vs ~0.5 GB HBM). fp32 emulated with bf16x3 on
// V_WMMA_F32_16X16X32_BF16 (split hoisted to a one-time pre-pass).
//   pass 0: split x/Wq/Wk/Wv into bf16 hi/lo arrays in workspace.
//   pass 1: k,v GEMM -> deterministic per-row partials of k*v and |k|.
//   pass 2: reduce partials -> kv[b], ksum[b].
//   pass 3: q GEMM + pointwise epilogue -> out.
// Staging: global_load_async_to_lds_b128 (ASYNCcnt), double-buffered LDS,
// software-pipelined so DMA overlaps WMMA. bf16 fragments are two
// ds_load_b128 each (lane half h holds k 16h..16h+15, same order for A and B).
// Workspace: ~352 MB.
// ---------------------------------------------------------------------------

typedef __attribute__((ext_vector_type(16))) __bf16        v16bf;
typedef __attribute__((ext_vector_type(8)))  float         v8f;
typedef __attribute__((ext_vector_type(8)))  unsigned int  v8u;

static constexpr int Mtot = 8192, Ktot = 4096, Ntot = 4096;
static constexpr int BM = 64, BN = 32, BK = 32;      // bf16 K-chunk = 16 dwords
static constexpr int Kdw = Ktot / 2;                 // global row pitch, dwords
static constexpr int RS  = 20;                       // LDS row stride, dwords
static constexpr int NGRP = Ntot / 16;               // 256 partial column-groups

#define USE_ASYNC_LDS 1
#define DEV __device__ __forceinline__

DEV unsigned int bf16_rne(float f) {
  unsigned int u = __builtin_bit_cast(unsigned int, f);
  return (u + 0x7FFFu + ((u >> 16) & 1u)) >> 16;
}
DEV float bf16_as_f32(unsigned int b) {
  return __builtin_bit_cast(float, b << 16);
}

// ---------------- pass 0: fp32 -> bf16 hi/lo split (off the hot path) ------
__global__ __launch_bounds__(256) void split_bf16_kernel(
    const float* __restrict__ src, unsigned int* __restrict__ hi,
    unsigned int* __restrict__ lo) {
  size_t i4 = (size_t)blockIdx.x * 256 + threadIdx.x;  // one float4 per thread
  float4 f = ((const float4*)src)[i4];
  unsigned int h0 = bf16_rne(f.x), h1 = bf16_rne(f.y);
  unsigned int h2 = bf16_rne(f.z), h3 = bf16_rne(f.w);
  unsigned int l0 = bf16_rne(f.x - bf16_as_f32(h0));
  unsigned int l1 = bf16_rne(f.y - bf16_as_f32(h1));
  unsigned int l2 = bf16_rne(f.z - bf16_as_f32(h2));
  unsigned int l3 = bf16_rne(f.w - bf16_as_f32(h3));
  uint2 ho; ho.x = h0 | (h1 << 16); ho.y = h2 | (h3 << 16);
  uint2 lv; lv.x = l0 | (l1 << 16); lv.y = l2 | (l3 << 16);
  ((uint2*)hi)[i4] = ho;
  ((uint2*)lo)[i4] = lv;
}

// ---------------- staging: global bf16 chunk -> padded LDS (async DMA) -----
// rows x 16-dword chunk; global pitch Kdw dwords, LDS pitch RS dwords.
DEV void stage_chunk(unsigned int* __restrict__ lds,
                     const unsigned int* __restrict__ gsrc,
                     int rows, unsigned int gdw0, int tid) {
  for (int idx = tid; idx < rows * 4; idx += 256) {
    int r = idx >> 2, c4 = idx & 3;
#if USE_ASYNC_LDS && defined(__gfx1250__)
    unsigned int goff = (gdw0 + (unsigned int)r * (unsigned int)Kdw +
                         (unsigned int)c4 * 4u) * 4u;            // bytes
    unsigned int loff =
        (unsigned int)(uintptr_t)(lds + r * RS + c4 * 4);        // LDS bytes
    asm volatile("global_load_async_to_lds_b128 %0, %1, %2"
                 :: "v"(loff), "v"(goff), "s"(gsrc)
                 : "memory");
#else
    uint4 val = *(const uint4*)(gsrc + gdw0 + (size_t)r * Kdw + c4 * 4);
    *(uint4*)(lds + r * RS + c4 * 4) = val;
#endif
  }
}

// Wait for this wave's outstanding DMA, then block-wide visibility.
DEV void stage_fence() {
#if USE_ASYNC_LDS && defined(__gfx1250__)
  asm volatile("s_wait_asynccnt 0x0" ::: "memory");
#endif
  __syncthreads();
}

// ---------------- fragments: two ds_load_b128 per v16bf --------------------
// A 16x32 (MxK): lanes 0-15 -> dwords {0..7};        lanes 16-31 -> {8..15}
// B 32x16 (KxN): lanes 0-15 -> dwords {0..7};        lanes 16-31 -> {8..15}
DEV v16bf frag_from(const unsigned int* rp, int d0, int d1) {
  uint4 p0 = *(const uint4*)(rp + d0);
  uint4 p1 = *(const uint4*)(rp + d1);
  v8u u;
  u[0] = p0.x; u[1] = p0.y; u[2] = p0.z; u[3] = p0.w;
  u[4] = p1.x; u[5] = p1.y; u[6] = p1.z; u[7] = p1.w;
  return __builtin_bit_cast(v16bf, u);
}

DEV v8f mma3(v8f c, v16bf ah, v16bf al, v16bf bh, v16bf bl) {
  c = __builtin_amdgcn_wmma_f32_16x16x32_bf16(false, ah, false, bh, (short)0, c, false, false);
  c = __builtin_amdgcn_wmma_f32_16x16x32_bf16(false, al, false, bh, (short)0, c, false, false);
  c = __builtin_amdgcn_wmma_f32_16x16x32_bf16(false, ah, false, bl, (short)0, c, false, false);
  return c;
}

// ---------------- pass 1: k,v tiles -> deterministic row partials ----------
__global__ __launch_bounds__(256) void kv_partials_kernel(
    const unsigned int* __restrict__ xh, const unsigned int* __restrict__ xl,
    const unsigned int* __restrict__ wkh, const unsigned int* __restrict__ wkl,
    const unsigned int* __restrict__ wvh, const unsigned int* __restrict__ wvl,
    const float* __restrict__ bk, const float* __restrict__ bv,
    float* __restrict__ pkv, float* __restrict__ pks) {
  __shared__ unsigned int xsh[2][BM * RS], xsl[2][BM * RS];
  __shared__ unsigned int skh[2][BN * RS], skl[2][BN * RS];
  __shared__ unsigned int svh[2][BN * RS], svl[2][BN * RS];
  const int n0 = blockIdx.x * BN, m0 = blockIdx.y * BM;
  const int t = threadIdx.x, lane = t & 31, wid = t >> 5;
  const int hl = lane >> 4, l16 = lane & 15;
  const int tr = wid >> 1, tcn = wid & 1;            // 4 M-tiles x 2 N-tiles
  const int tx = t ^ 128;                            // balance async issue

  v8f acc_k = {}, acc_v = {};
  const unsigned int* xhg = xh + (size_t)m0 * Kdw;
  const unsigned int* xlg = xl + (size_t)m0 * Kdw;
  const unsigned int* khg = wkh + (size_t)n0 * Kdw;
  const unsigned int* klg = wkl + (size_t)n0 * Kdw;
  const unsigned int* vhg = wvh + (size_t)n0 * Kdw;
  const unsigned int* vlg = wvl + (size_t)n0 * Kdw;

  // prologue: chunk 0 -> buffer 0
  stage_chunk(xsh[0], xhg, BM, 0, t);
  stage_chunk(xsl[0], xlg, BM, 0, tx);
  stage_chunk(skh[0], khg, BN, 0, t);
  stage_chunk(skl[0], klg, BN, 0, tx);
  stage_chunk(svh[0], vhg, BN, 0, t);
  stage_chunk(svl[0], vlg, BN, 0, tx);

  for (int kc = 0; kc < Ktot; kc += BK) {
    const int cur = (kc / BK) & 1;
    stage_fence();                      // chunk kc landed everywhere
    if (kc + BK < Ktot) {               // overlap next DMA with this compute
      unsigned int gn = (unsigned int)((kc + BK) / 2);
      stage_chunk(xsh[cur ^ 1], xhg, BM, gn, t);
      stage_chunk(xsl[cur ^ 1], xlg, BM, gn, tx);
      stage_chunk(skh[cur ^ 1], khg, BN, gn, t);
      stage_chunk(skl[cur ^ 1], klg, BN, gn, tx);
      stage_chunk(svh[cur ^ 1], vhg, BN, gn, t);
      stage_chunk(svl[cur ^ 1], vlg, BN, gn, tx);
    }
    __builtin_prefetch(xhg + (size_t)(kc / 2) + 32 + (size_t)(t & 63) * Kdw, 0, 0);

    const unsigned int* ar_h = &xsh[cur][(tr * 16 + l16) * RS];
    const unsigned int* ar_l = &xsl[cur][(tr * 16 + l16) * RS];
    v16bf ah = frag_from(ar_h, hl * 8, hl * 8 + 4);
    v16bf al = frag_from(ar_l, hl * 8, hl * 8 + 4);
    const unsigned int* bk_h = &skh[cur][(tcn * 16 + l16) * RS];
    const unsigned int* bk_l = &skl[cur][(tcn * 16 + l16) * RS];
    v16bf bh = frag_from(bk_h, hl * 8, hl * 8 + 4);
    v16bf bl = frag_from(bk_l, hl * 8, hl * 8 + 4);
    acc_k = mma3(acc_k, ah, al, bh, bl);
    const unsigned int* bv_h = &svh[cur][(tcn * 16 + l16) * RS];
    const unsigned int* bv_l = &svl[cur][(tcn * 16 + l16) * RS];
    bh = frag_from(bv_h, hl * 8, hl * 8 + 4);
    bl = frag_from(bv_l, hl * 8, hl * 8 + 4);
    acc_v = mma3(acc_v, ah, al, bh, bl);
  }

  // per-row partial sums over this tile's 16 N-columns (C layout: N = lane%16)
  const int ng = n0 + tcn * 16 + l16;
  const float bkv = bk[ng], bvv = bv[ng];
  const int grp = blockIdx.x * 2 + tcn;              // N column-group, 0..255
#pragma unroll
  for (int i = 0; i < 8; ++i) {
    float kval = acc_k[i] + bkv;
    float vval = acc_v[i] + bvv;
    float e = kval * vval;
    float a = __builtin_fabsf(kval);
#pragma unroll
    for (int off = 8; off > 0; off >>= 1) {          // reduce 16-lane N group
      e += __shfl_xor(e, off, 32);
      a += __shfl_xor(a, off, 32);
    }
    if (l16 == 0) {
      int m = m0 + tr * 16 + i + hl * 8;
      pkv[(size_t)grp * Mtot + m] = e;               // race-free: unique (grp,m)
      pks[(size_t)grp * Mtot + m] = a;
    }
  }
}

// ---------------- pass 2: reduce partials -> kv[b], ksum[b] ----------------
__global__ __launch_bounds__(256) void reduce_kernel(
    const float* __restrict__ pkv, const float* __restrict__ pks,
    float* __restrict__ kv, float* __restrict__ ks) {
  int m = blockIdx.x * 256 + threadIdx.x;
  float skv = 0.f, sks = 0.f;
  for (int p = 0; p < NGRP; ++p) {
    skv += pkv[(size_t)p * Mtot + m];
    sks += pks[(size_t)p * Mtot + m];
  }
  kv[m] = skv;
  ks[m] = sks;
}

// ---------------- pass 3: q GEMM + pointwise epilogue -> out ---------------
__global__ __launch_bounds__(256) void q_out_kernel(
    const unsigned int* __restrict__ xh, const unsigned int* __restrict__ xl,
    const unsigned int* __restrict__ wqh, const unsigned int* __restrict__ wql,
    const float* __restrict__ bq, const float* __restrict__ kv,
    const float* __restrict__ ks, float* __restrict__ out) {
  __shared__ unsigned int xsh[2][BM * RS], xsl[2][BM * RS];
  __shared__ unsigned int sqh[2][BN * RS], sql[2][BN * RS];
  const int n0 = blockIdx.x * BN, m0 = blockIdx.y * BM;
  const int t = threadIdx.x, lane = t & 31, wid = t >> 5;
  const int hl = lane >> 4, l16 = lane & 15;
  const int tr = wid >> 1, tcn = wid & 1;
  const int tx = t ^ 128;

  v8f acc = {};
  const unsigned int* xhg = xh + (size_t)m0 * Kdw;
  const unsigned int* xlg = xl + (size_t)m0 * Kdw;
  const unsigned int* qhg = wqh + (size_t)n0 * Kdw;
  const unsigned int* qlg = wql + (size_t)n0 * Kdw;

  stage_chunk(xsh[0], xhg, BM, 0, t);
  stage_chunk(xsl[0], xlg, BM, 0, tx);
  stage_chunk(sqh[0], qhg, BN, 0, t);
  stage_chunk(sql[0], qlg, BN, 0, tx);

  for (int kc = 0; kc < Ktot; kc += BK) {
    const int cur = (kc / BK) & 1;
    stage_fence();
    if (kc + BK < Ktot) {
      unsigned int gn = (unsigned int)((kc + BK) / 2);
      stage_chunk(xsh[cur ^ 1], xhg, BM, gn, t);
      stage_chunk(xsl[cur ^ 1], xlg, BM, gn, tx);
      stage_chunk(sqh[cur ^ 1], qhg, BN, gn, t);
      stage_chunk(sql[cur ^ 1], qlg, BN, gn, tx);
    }
    __builtin_prefetch(xhg + (size_t)(kc / 2) + 32 + (size_t)(t & 63) * Kdw, 0, 0);

    const unsigned int* ar_h = &xsh[cur][(tr * 16 + l16) * RS];
    const unsigned int* ar_l = &xsl[cur][(tr * 16 + l16) * RS];
    v16bf ah = frag_from(ar_h, hl * 8, hl * 8 + 4);
    v16bf al = frag_from(ar_l, hl * 8, hl * 8 + 4);
    const unsigned int* bq_h = &sqh[cur][(tcn * 16 + l16) * RS];
    const unsigned int* bq_l = &sql[cur][(tcn * 16 + l16) * RS];
    v16bf bh = frag_from(bq_h, hl * 8, hl * 8 + 4);
    v16bf bl = frag_from(bq_l, hl * 8, hl * 8 + 4);
    acc = mma3(acc, ah, al, bh, bl);
  }

  const int ng = n0 + tcn * 16 + l16;
  const float bqv = bq[ng];
#pragma unroll
  for (int i = 0; i < 8; ++i) {
    int m = m0 + tr * 16 + i + hl * 8;
    float q = acc[i] + bqv;
    float r = q * kv[m] / (1.0f + __builtin_fabsf(q) * ks[m]);
    out[(size_t)m * Ntot + ng] = r;
  }
}

// ---------------------------------------------------------------------------
extern "C" void kernel_launch(void* const* d_in, const int* in_sizes, int n_in,
                              void* d_out, int out_size, void* d_ws, size_t ws_size,
                              hipStream_t stream) {
  const float* x  = (const float*)d_in[0];
  const float* Wq = (const float*)d_in[1];
  const float* bq = (const float*)d_in[2];
  const float* Wk = (const float*)d_in[3];
  const float* bk = (const float*)d_in[4];
  const float* Wv = (const float*)d_in[5];
  const float* bv = (const float*)d_in[6];
  float* out = (float*)d_out;

  // Workspace layout (fully rewritten every launch -> poison-safe):
  const size_t szX = (size_t)Mtot * Ktot;   // elements
  const size_t szW = (size_t)Ntot * Ktot;
  char* w = (char*)d_ws;
  unsigned int* xh  = (unsigned int*)w;                 w += szX * 2;   // bf16
  unsigned int* xl  = (unsigned int*)w;                 w += szX * 2;
  unsigned int* wqh = (unsigned int*)w;                 w += szW * 2;
  unsigned int* wql = (unsigned int*)w;                 w += szW * 2;
  unsigned int* wkh = (unsigned int*)w;                 w += szW * 2;
  unsigned int* wkl = (unsigned int*)w;                 w += szW * 2;
  unsigned int* wvh = (unsigned int*)w;                 w += szW * 2;
  unsigned int* wvl = (unsigned int*)w;                 w += szW * 2;
  float* pkv = (float*)w;                               w += (size_t)NGRP * Mtot * 4;
  float* pks = (float*)w;                               w += (size_t)NGRP * Mtot * 4;
  float* kv  = (float*)w;                               w += (size_t)Mtot * 4;
  float* ks  = (float*)w;

  // pass 0: fp32 -> bf16 hi/lo splits
  split_bf16_kernel<<<dim3((unsigned)(szX / 4 / 256)), 256, 0, stream>>>(x,  xh,  xl);
  split_bf16_kernel<<<dim3((unsigned)(szW / 4 / 256)), 256, 0, stream>>>(Wq, wqh, wql);
  split_bf16_kernel<<<dim3((unsigned)(szW / 4 / 256)), 256, 0, stream>>>(Wk, wkh, wkl);
  split_bf16_kernel<<<dim3((unsigned)(szW / 4 / 256)), 256, 0, stream>>>(Wv, wvh, wvl);

  dim3 grid(Ntot / BN, Mtot / BM);                      // 128 x 128
  kv_partials_kernel<<<grid, 256, 0, stream>>>(xh, xl, wkh, wkl, wvh, wvl,
                                               bk, bv, pkv, pks);
  reduce_kernel<<<dim3(Mtot / 256), 256, 0, stream>>>(pkv, pks, kv, ks);
  q_out_kernel<<<grid, 256, 0, stream>>>(xh, xl, wqh, wql, bq, kv, ks, out);
}
